// LlamaAttention_64544768524702
// MI455X (gfx1250) — matrix operand order repaired and checked by reading, not compiled, not run on hardware
//
#include <hip/hip_runtime.h>
#include <hip/hip_bf16.h>
#include <math.h>

typedef __attribute__((ext_vector_type(16))) __bf16 v16bf;
typedef __attribute__((ext_vector_type(8)))  __bf16 bf16x8;
typedef __attribute__((ext_vector_type(8)))  float  v8f;

#define TT   4096   // total tokens
#define DD   4096   // hidden
#define NH   32     // q heads
#define KH   8      // kv heads
#define HD   128    // head dim
#define BB   4      // sequences
#define SS   1024   // seq len
#define QKVN 6144   // NH*HD + 2*KH*HD

// ---------------------------------------------------------------------------
// fp32 -> bf16 cast with column placement (for packing Wq|Wk|Wv side by side)
// ---------------------------------------------------------------------------
__global__ void cast_pack_kernel(const float* __restrict__ src, __bf16* __restrict__ dst,
                                 int rows, int src_cols, int dst_stride, int col_off) {
  int idx = blockIdx.x * blockDim.x + threadIdx.x;
  int total = rows * src_cols;
  if (idx >= total) return;
  int r = idx / src_cols;
  int c = idx - r * src_cols;
  dst[(size_t)r * dst_stride + col_off + c] = (__bf16)src[idx];
}

// ---------------------------------------------------------------------------
// Tiled WMMA GEMM: C[M,N] = A[M,K] * B[K,N], A/B bf16 row-major, fp32 acc.
// Block tile 128x128, 8 wave32 waves (4x2), wave tile 32x64 (2x4 WMMA tiles).
// Double-buffered LDS with register-staged global prefetch: ONE barrier/iter,
// next tile's global loads issue before the barrier so HBM/L2 latency is
// hidden behind the current tile's WMMAs. B tile is stored TRANSPOSED in LDS
// so both A and B fragment loads are contiguous ds_load_b128 pairs.
// ---------------------------------------------------------------------------
template<int OUT_BF16>
__global__ __launch_bounds__(256)
void gemm_bf16_kernel(const __bf16* __restrict__ A, const __bf16* __restrict__ B,
                      float* __restrict__ Cf, __bf16* __restrict__ Cb,
                      int N, int K) {
  __shared__ __align__(16) __bf16 As[2][128][48];  // [buf][m][k], 96B rows
  __shared__ __align__(16) __bf16 Bt[2][128][48];  // [buf][n][k] (transposed), 96B rows

  const int tid  = threadIdx.x;
  const int lane = tid & 31;
  const int wid  = tid >> 5;
  const int half = lane >> 4;
  const int lr   = lane & 15;
  const int wm   = wid >> 1;            // 0..3
  const int wn   = wid & 1;             // 0..1
  const int bm   = blockIdx.y * 128;
  const int bn   = blockIdx.x * 128;

  // per-thread staging: A tile 128x32 -> 2 vec8 / thread; B tile 32x128 -> 2 vec8
  const int ar0 = tid >> 2,  ac0 = (tid & 3) << 3;          // A chunk 0: row, col
  const int ar1 = (tid + 256) >> 2, ac1 = ((tid + 256) & 3) << 3;
  const int br0 = tid >> 4,  bc0 = (tid & 15) << 3;         // B chunk 0: krow, ncol
  const int br1 = (tid + 256) >> 4, bc1 = ((tid + 256) & 15) << 3;

  v8f acc[2][4] = {};
  bf16x8 ra0, ra1, rb0, rb1;

  // prologue: fetch tile 0
  ra0 = *(const bf16x8*)(&A[(size_t)(bm + ar0) * K + ac0]);
  ra1 = *(const bf16x8*)(&A[(size_t)(bm + ar1) * K + ac1]);
  rb0 = *(const bf16x8*)(&B[(size_t)br0 * N + bn + bc0]);
  rb1 = *(const bf16x8*)(&B[(size_t)br1 * N + bn + bc1]);

  int buf = 0;
  for (int k0 = 0; k0 < K; k0 += 32) {
    // stash staged tile into LDS (B transposed)
    *(bf16x8*)(&As[buf][ar0][ac0]) = ra0;
    *(bf16x8*)(&As[buf][ar1][ac1]) = ra1;
    #pragma unroll
    for (int j = 0; j < 8; ++j) Bt[buf][bc0 + j][br0] = rb0[j];
    #pragma unroll
    for (int j = 0; j < 8; ++j) Bt[buf][bc1 + j][br1] = rb1[j];

    // issue next tile's global loads before the barrier (latency overlap)
    bf16x8 na0 = {}, na1 = {}, nb0 = {}, nb1 = {};
    const int kn = k0 + 32;
    if (kn < K) {
      na0 = *(const bf16x8*)(&A[(size_t)(bm + ar0) * K + kn + ac0]);
      na1 = *(const bf16x8*)(&A[(size_t)(bm + ar1) * K + kn + ac1]);
      nb0 = *(const bf16x8*)(&B[(size_t)(kn + br0) * N + bn + bc0]);
      nb1 = *(const bf16x8*)(&B[(size_t)(kn + br1) * N + bn + bc1]);
    }
    if (k0 + 64 < K) {  // warm L2 two tiles ahead (global_prefetch_b8)
      __builtin_prefetch(&A[(size_t)(bm + ar0) * K + k0 + 64 + ac0], 0, 1);
      __builtin_prefetch(&B[(size_t)(k0 + 64 + br0) * N + bn + bc0], 0, 1);
    }

    __syncthreads();

    const __bf16 (*as)[48] = As[buf];
    const __bf16 (*bs)[48] = Bt[buf];
    v16bf af[2], bfr[4];
    #pragma unroll
    for (int mt = 0; mt < 2; ++mt) {
      int row = wm * 32 + mt * 16 + lr;
      #pragma unroll
      for (int e = 0; e < 16; ++e) {
        int kk = e + (half << 4);                            // A-frag 16x32 layout
        af[mt][e] = as[row][kk];
      }
    }
    #pragma unroll
    for (int nt = 0; nt < 4; ++nt) {
      int col = wn * 64 + nt * 16 + lr;
      #pragma unroll
      for (int e = 0; e < 16; ++e) {
        int kk = e + (half << 4);                            // B-frag 32x16 layout
        bfr[nt][e] = bs[col][kk];                            // contiguous -> ds_load_b128
      }
    }
    #pragma unroll
    for (int mt = 0; mt < 2; ++mt)
      #pragma unroll
      for (int nt = 0; nt < 4; ++nt)
        acc[mt][nt] = __builtin_amdgcn_wmma_f32_16x16x32_bf16(
            false, af[mt], false, bfr[nt], (short)0, acc[mt][nt], false, false);

    ra0 = na0; ra1 = na1; rb0 = nb0; rb1 = nb1;
    buf ^= 1;
  }

  // Epilogue: C layout row m = v + 8*half, col = lane%16
  #pragma unroll
  for (int mt = 0; mt < 2; ++mt) {
    #pragma unroll
    for (int nt = 0; nt < 4; ++nt) {
      int coln = bn + wn * 64 + nt * 16 + lr;
      #pragma unroll
      for (int v = 0; v < 8; ++v) {
        int rowm = bm + wm * 32 + mt * 16 + v + (half << 3);
        if (OUT_BF16) Cb[(size_t)rowm * N + coln] = (__bf16)acc[mt][nt][v];
        else          Cf[(size_t)rowm * N + coln] = acc[mt][nt][v];
      }
    }
  }
}

// ---------------------------------------------------------------------------
// RoPE on q and k (fp32 math), reading fused qkv bf16, writing bf16.
// out[i] = x[i]*cos - x[i+64]*sin ; out[i+64] = x[i+64]*cos + x[i]*sin
// ---------------------------------------------------------------------------
__global__ void rope_kernel(const __bf16* __restrict__ qkv, const int* __restrict__ pos,
                            __bf16* __restrict__ qo, __bf16* __restrict__ ko) {
  int idx = blockIdx.x * blockDim.x + threadIdx.x;
  const int total = TT * (NH + KH) * (HD / 2);
  if (idx >= total) return;
  int i = idx & 63;
  int h = (idx >> 6) % (NH + KH);
  int t = idx / (64 * (NH + KH));
  float p    = (float)pos[t];
  float freq = __expf(-(2.0f * (float)i / 128.0f) * 9.210340371976184f); // 10000^(-2i/H)
  float ang  = p * freq;
  float cs = cosf(ang), sn = sinf(ang);
  if (h < NH) {
    size_t base = (size_t)t * QKVN + h * HD;
    float x1 = (float)qkv[base + i];
    float x2 = (float)qkv[base + 64 + i];
    size_t ob = (size_t)t * (NH * HD) + h * HD;
    qo[ob + i]      = (__bf16)(x1 * cs - x2 * sn);
    qo[ob + 64 + i] = (__bf16)(x2 * cs + x1 * sn);
  } else {
    int kh = h - NH;
    size_t base = (size_t)t * QKVN + NH * HD + kh * HD;
    float x1 = (float)qkv[base + i];
    float x2 = (float)qkv[base + 64 + i];
    size_t ob = (size_t)t * (KH * HD) + kh * HD;
    ko[ob + i]      = (__bf16)(x1 * cs - x2 * sn);
    ko[ob + 64 + i] = (__bf16)(x2 * cs + x1 * sn);
  }
}

__global__ void vcopy_kernel(const __bf16* __restrict__ qkv, __bf16* __restrict__ vb) {
  int idx = blockIdx.x * blockDim.x + threadIdx.x;
  if (idx >= TT * KH * HD) return;
  int t = idx / (KH * HD);
  int c = idx - t * (KH * HD);
  vb[idx] = qkv[(size_t)t * QKVN + NH * HD + KH * HD + c];
}

// ---------------------------------------------------------------------------
// Flash attention, causal, GQA. Block = one (seq, q-head, 64-row q tile).
// 4 wave32 waves; each wave owns 16 q rows. Key chunks of 32, online softmax.
// Double-buffered K/V chunks with register staging: ONE barrier per chunk.
// ---------------------------------------------------------------------------
__global__ __launch_bounds__(128)
void fa_kernel(const __bf16* __restrict__ Q, const __bf16* __restrict__ Kr,
               const __bf16* __restrict__ V, __bf16* __restrict__ O) {
  __shared__ __align__(16) __bf16 Qs[64][128];
  __shared__ __align__(16) __bf16 Ks[2][32][136];
  __shared__ __align__(16) __bf16 Vs[2][32][136];
  __shared__ __align__(16) __bf16 Ps[4][16][32];

  const int tid  = threadIdx.x;
  const int lane = tid & 31;
  const int wid  = tid >> 5;      // 0..3
  const int half = lane >> 4;
  const int lr   = lane & 15;

  const int qt    = blockIdx.x;   // 0..15 (q tile of 64 rows)
  const int n     = blockIdx.y;   // q head
  const int b     = blockIdx.z;   // sequence
  const int kh    = n >> 2;       // kv head (G = 4)
  const int qrow0 = qt * 64;      // seq-local
  const int trow0 = b * SS + qrow0;

  // stage Q tile 64x128
  for (int i = tid; i < 64 * 16; i += 128) {
    int r = i >> 4, c8 = (i & 15) << 3;
    *(bf16x8*)(&Qs[r][c8]) =
        *(const bf16x8*)(&Q[(size_t)(trow0 + r) * (NH * HD) + n * HD + c8]);
  }
  __syncthreads();

  // Q A-fragments for the 4 H-chunks of 32
  v16bf qf[4];
  #pragma unroll
  for (int kc = 0; kc < 4; ++kc) {
    int row = wid * 16 + lr;
    #pragma unroll
    for (int e = 0; e < 16; ++e) {
      int kk = kc * 32 + e + (half << 4);
      qf[kc][e] = Qs[row][kk];
    }
  }

  float mrun[8], lrun[8];
  v8f oacc[8] = {};
  #pragma unroll
  for (int v = 0; v < 8; ++v) { mrun[v] = -3.0e38f; lrun[v] = 0.f; }

  const int   nchunks = 2 * qt + 2;            // causal chunk skipping
  const float scale   = 0.08838834764831845f;  // 1/sqrt(128)

  // per-thread KV staging: 32x128 chunk -> 4 vec8 per matrix per thread
  int rr[4], cc[4];
  #pragma unroll
  for (int u = 0; u < 4; ++u) {
    int i = tid + u * 128;
    rr[u] = i >> 4;
    cc[u] = (i & 15) << 3;
  }
  bf16x8 rk[4], rv[4];
  #pragma unroll
  for (int u = 0; u < 4; ++u) {
    size_t src = (size_t)(b * SS + rr[u]) * (KH * HD) + kh * HD + cc[u];
    rk[u] = *(const bf16x8*)(&Kr[src]);
    rv[u] = *(const bf16x8*)(&V[src]);
  }

  int buf = 0;
  for (int c0 = 0; c0 < nchunks; ++c0) {
    const int kb = c0 * 32;
    // stash current chunk
    #pragma unroll
    for (int u = 0; u < 4; ++u) {
      *(bf16x8*)(&Ks[buf][rr[u]][cc[u]]) = rk[u];
      *(bf16x8*)(&Vs[buf][rr[u]][cc[u]]) = rv[u];
    }
    // issue next chunk's global loads before the barrier
    bf16x8 nk[4] = {}, nv[4] = {};
    if (c0 + 1 < nchunks) {
      #pragma unroll
      for (int u = 0; u < 4; ++u) {
        size_t src = (size_t)(b * SS + kb + 32 + rr[u]) * (KH * HD) + kh * HD + cc[u];
        nk[u] = *(const bf16x8*)(&Kr[src]);
        nv[u] = *(const bf16x8*)(&V[src]);
      }
    }
    __syncthreads();

    // scores: 16 q rows x 32 keys
    v8f sc[2] = {};
    #pragma unroll
    for (int nt = 0; nt < 2; ++nt) {
      #pragma unroll
      for (int kc = 0; kc < 4; ++kc) {
        v16bf bfr;
        #pragma unroll
        for (int e = 0; e < 16; ++e) {
          int kk = kc * 32 + e + (half << 4);      // B row index = H-dim (K^T)
          bfr[e] = Ks[buf][nt * 16 + lr][kk];
        }
        sc[nt] = __builtin_amdgcn_wmma_f32_16x16x32_bf16(
            false, qf[kc], false, bfr, (short)0, sc[nt], false, false);
      }
    }
    // scale + causal mask
    #pragma unroll
    for (int nt = 0; nt < 2; ++nt) {
      int key = kb + nt * 16 + lr;
      #pragma unroll
      for (int v = 0; v < 8; ++v) {
        int qp = qrow0 + wid * 16 + v + (half << 3);
        float s = sc[nt][v] * scale;
        sc[nt][v] = (key > qp) ? -3.0e38f : s;
      }
    }
    // online softmax: vgpr v holds row v (lanes 0-15) / row v+8 (lanes 16-31)
    #pragma unroll
    for (int v = 0; v < 8; ++v) {
      float mx = fmaxf(sc[0][v], sc[1][v]);
      #pragma unroll
      for (int off = 1; off < 16; off <<= 1) mx = fmaxf(mx, __shfl_xor(mx, off, 32));
      float mnew  = fmaxf(mrun[v], mx);
      float alpha = __expf(mrun[v] - mnew);
      float p0 = __expf(sc[0][v] - mnew);
      float p1 = __expf(sc[1][v] - mnew);
      float rs = p0 + p1;
      #pragma unroll
      for (int off = 1; off < 16; off <<= 1) rs += __shfl_xor(rs, off, 32);
      lrun[v] = lrun[v] * alpha + rs;
      mrun[v] = mnew;
      #pragma unroll
      for (int ot = 0; ot < 8; ++ot) oacc[ot][v] *= alpha;
      Ps[wid][v + (half << 3)][lr]      = (__bf16)p0;
      Ps[wid][v + (half << 3)][16 + lr] = (__bf16)p1;
    }
    // reload P in A-fragment layout (rows = q, K-dim = 32 keys)
    v16bf pf;
    #pragma unroll
    for (int e = 0; e < 16; ++e) {
      int kk = e + (half << 4);
      pf[e] = Ps[wid][lr][kk];
    }
    // PV: accumulate 16x128 output
    #pragma unroll
    for (int ot = 0; ot < 8; ++ot) {
      v16bf vf;
      #pragma unroll
      for (int e = 0; e < 16; ++e) {
        int kk = e + (half << 4);                  // B row = key index
        vf[e] = Vs[buf][kk][ot * 16 + lr];
      }
      oacc[ot] = __builtin_amdgcn_wmma_f32_16x16x32_bf16(
          false, pf, false, vf, (short)0, oacc[ot], false, false);
    }

    #pragma unroll
    for (int u = 0; u < 4; ++u) { rk[u] = nk[u]; rv[u] = nv[u]; }
    buf ^= 1;
  }

  // normalize + write attn_out bf16 [T, N*H]
  #pragma unroll
  for (int v = 0; v < 8; ++v) {
    float inv = 1.0f / lrun[v];
    int t = trow0 + wid * 16 + v + (half << 3);
    #pragma unroll
    for (int ot = 0; ot < 8; ++ot) {
      O[(size_t)t * (NH * HD) + n * HD + ot * 16 + lr] = (__bf16)(oacc[ot][v] * inv);
    }
  }
}

// ---------------------------------------------------------------------------
extern "C" void kernel_launch(void* const* d_in, const int* in_sizes, int n_in,
                              void* d_out, int out_size, void* d_ws, size_t ws_size,
                              hipStream_t stream) {
  const float* x    = (const float*)d_in[0];
  const int*   posi = (const int*)d_in[1];
  const float* Wq   = (const float*)d_in[2];
  const float* Wk   = (const float*)d_in[3];
  const float* Wv   = (const float*)d_in[4];
  const float* Wo   = (const float*)d_in[5];
  float* out = (float*)d_out;

  char* w = (char*)d_ws;
  auto carve = [&](size_t bytes) -> char* {
    char* p = w;
    w += (bytes + 255) & ~(size_t)255;
    return p;
  };
  __bf16* x_bf  = (__bf16*)carve((size_t)TT * DD * 2);
  __bf16* wqkv  = (__bf16*)carve((size_t)DD * QKVN * 2);
  __bf16* wo_bf = (__bf16*)carve((size_t)DD * DD * 2);
  __bf16* qkv   = (__bf16*)carve((size_t)TT * QKVN * 2);
  __bf16* q_r   = (__bf16*)carve((size_t)TT * NH * HD * 2);
  __bf16* k_r   = (__bf16*)carve((size_t)TT * KH * HD * 2);
  __bf16* v_b   = (__bf16*)carve((size_t)TT * KH * HD * 2);
  __bf16* a_o   = (__bf16*)carve((size_t)TT * NH * HD * 2);

  const int thr = 256;
  cast_pack_kernel<<<(TT * DD + thr - 1) / thr, thr, 0, stream>>>(x, x_bf, TT, DD, DD, 0);
  cast_pack_kernel<<<(DD * 4096 + thr - 1) / thr, thr, 0, stream>>>(Wq, wqkv, DD, 4096, QKVN, 0);
  cast_pack_kernel<<<(DD * 1024 + thr - 1) / thr, thr, 0, stream>>>(Wk, wqkv, DD, 1024, QKVN, 4096);
  cast_pack_kernel<<<(DD * 1024 + thr - 1) / thr, thr, 0, stream>>>(Wv, wqkv, DD, 1024, QKVN, 5120);
  cast_pack_kernel<<<(DD * DD + thr - 1) / thr, thr, 0, stream>>>(Wo, wo_bf, DD, DD, DD, 0);

  // QKV projection: [T,D] x [D,6144] -> bf16 qkv
  gemm_bf16_kernel<1><<<dim3(QKVN / 128, TT / 128), 256, 0, stream>>>(
      x_bf, wqkv, nullptr, qkv, QKVN, DD);

  rope_kernel<<<(TT * (NH + KH) * 64 + thr - 1) / thr, thr, 0, stream>>>(qkv, posi, q_r, k_r);
  vcopy_kernel<<<(TT * KH * HD + thr - 1) / thr, thr, 0, stream>>>(qkv, v_b);

  // causal GQA flash attention
  fa_kernel<<<dim3(SS / 64, NH, BB), 128, 0, stream>>>(q_r, k_r, v_b, a_o);

  // output projection: [T,4096] x [4096,4096] -> fp32 d_out
  gemm_bf16_kernel<0><<<dim3(DD / 128, TT / 128), 256, 0, stream>>>(
      a_o, wo_bf, out, nullptr, DD, DD);
}
